// PairwiseRowAttention_22041772163330
// MI455X (gfx1250) — matrix operand order repaired and checked by reading, not compiled, not run on hardware
//
#include <hip/hip_runtime.h>

// ---------------------------------------------------------------------------
// PairwiseRowAttention fused kernel for MI455X (gfx1250, wave32, WMMA)
//
// One block per n (512 blocks, 256 threads = 8 waves).
//   Phase A: K = Xn@Wk, V = Xn@Wv + b -> LDS f16 (V transposed); Wq^T staged.
//   Phase B: per 16-row i-tile flash attention:
//            Q = Xn@Wq (WMMA + LDS transpose stage), online softmax over j
//            in steps of 32, O += P@V, normalize, store.
//
// LDS: K[512][128] (128KB) + V^T[128][512] (128KB) + Wq^T[128][128] (32KB)
//      + 8 x per-wave 16x32 stage (8KB) = 296KB dynamic shared.
// All fragment reads from LDS are 16B vector loads (ds_load_b128).
// ---------------------------------------------------------------------------

typedef __attribute__((ext_vector_type(16))) _Float16 v16h;
typedef __attribute__((ext_vector_type(8)))  _Float16 v8h;
typedef __attribute__((ext_vector_type(8)))  float    v8f;

#define N_SEQ 512
#define D_DIM 128

// LDS offsets (in halfs)
#define K_OFF   0
#define VT_OFF  (N_SEQ * D_DIM)                   // 65536
#define WQT_OFF (VT_OFF + D_DIM * N_SEQ)          // 131072
#define STG_OFF (WQT_OFF + D_DIM * D_DIM)         // 147456
#define SMEM_HALFS (STG_OFF + 8 * 16 * 32)        // 151552 -> 296 KB

extern __shared__ _Float16 smem[];

__device__ __forceinline__ v8f wmma_f16(v16h a, v16h b, v8f c) {
  return __builtin_amdgcn_wmma_f32_16x16x32_f16(
      false, a, false, b, (short)0, c, false, false);
}

// 16 contiguous halfs from LDS -> one fragment (2x ds_load_b128)
__device__ __forceinline__ v16h ld_frag_lds(int idx) {
  const v8h lo = *(const v8h*)&smem[idx];
  const v8h hi = *(const v8h*)&smem[idx + 8];
  return __builtin_shufflevector(lo, hi, 0, 1, 2, 3, 4, 5, 6, 7,
                                 8, 9, 10, 11, 12, 13, 14, 15);
}

// A-fragment from a per-wave 16x32 stage buffer (row-major halfs)
__device__ __forceinline__ v16h ld_a_stage(int stg_base, int l16, int hh) {
  const v8h lo = *(const v8h*)&smem[stg_base + l16 * 32 + hh * 16];
  const v8h hi = *(const v8h*)&smem[stg_base + l16 * 32 + hh * 16 + 8];
  return __builtin_shufflevector(lo, hi, 0, 1, 2, 3, 4, 5, 6, 7,
                                 8, 9, 10, 11, 12, 13, 14, 15);
}

// A-fragment of X (f32 global, row-major, 32-column chunk at `row`)
__device__ __forceinline__ v16h ld_a_x(const float* __restrict__ row, int hh) {
  const float4 f0 = *(const float4*)(row + hh * 16);
  const float4 f1 = *(const float4*)(row + hh * 16 + 4);
  const float4 f2 = *(const float4*)(row + hh * 16 + 8);
  const float4 f3 = *(const float4*)(row + hh * 16 + 12);
  v16h a;
  a[0]  = (_Float16)f0.x; a[1]  = (_Float16)f0.y;
  a[2]  = (_Float16)f0.z; a[3]  = (_Float16)f0.w;
  a[4]  = (_Float16)f1.x; a[5]  = (_Float16)f1.y;
  a[6]  = (_Float16)f1.z; a[7]  = (_Float16)f1.w;
  a[8]  = (_Float16)f2.x; a[9]  = (_Float16)f2.y;
  a[10] = (_Float16)f2.z; a[11] = (_Float16)f2.w;
  a[12] = (_Float16)f3.x; a[13] = (_Float16)f3.y;
  a[14] = (_Float16)f3.z; a[15] = (_Float16)f3.w;
  return a;
}

__global__ __launch_bounds__(256, 1)
void pairwise_row_attention_kernel(const float* __restrict__ x,
                                   const float* __restrict__ w_qk,
                                   const float* __restrict__ w_v,
                                   const float* __restrict__ b_v,
                                   float* __restrict__ out) {
  const int n    = blockIdx.x;
  const int tid  = threadIdx.x;
  const int wave = tid >> 5;
  const int lane = tid & 31;
  const int hh   = lane >> 4;   // half-wave id (0/1)
  const int l16  = lane & 15;
  const float* xn = x + (size_t)n * N_SEQ * D_DIM;
  const int stg = STG_OFF + wave * (16 * 32);

  // ---- stage Wq TRANSPOSED into LDS: WQT[c][k] = w_qk[k][c], c,k in [0,128)
  // Coalesced global reads (consecutive tid -> consecutive c), LDS scatter.
  for (int idx = tid; idx < D_DIM * D_DIM; idx += 256) {
    const int k = idx >> 7, c = idx & 127;
    smem[WQT_OFF + c * D_DIM + k] = (_Float16)w_qk[k * 256 + c];
  }

  // =========================================================================
  // Phase A: K = Xn @ Wk, V = Xn @ Wv + b.  Wave w owns output col-tile w.
  // =========================================================================
  {
    const int nt   = wave;
    const int colK = 128 + nt * 16 + l16;   // Wk = cols 128..255 of w_qk
    const int colV = nt * 16 + l16;
    const float bias = b_v[colV];

    // One-time strided weight gathers (L2-resident), held in registers.
    v16h bk[4], bv[4];
#pragma unroll
    for (int kc = 0; kc < 4; ++kc) {
      const int kbase = kc * 32 + hh * 16;
#pragma unroll
      for (int e = 0; e < 16; ++e) {
        bk[kc][e] = (_Float16)w_qk[(kbase + e) * 256 + colK];
        bv[kc][e] = (_Float16)w_v [(kbase + e) * 128 + colV];
      }
    }

#pragma unroll 1
    for (int mt = 0; mt < 32; ++mt) {
      const int m0 = mt * 16;
      v8f ck = {}; v8f cv = {};
#pragma unroll
      for (int kc = 0; kc < 4; ++kc) {
        const v16h a = ld_a_x(xn + (size_t)(m0 + l16) * D_DIM + kc * 32, hh);
        ck = wmma_f16(a, bk[kc], ck);
        cv = wmma_f16(a, bv[kc], cv);
      }
#pragma unroll
      for (int r = 0; r < 8; ++r) {
        const int row = m0 + hh * 8 + r;
        smem[K_OFF + row * D_DIM + nt * 16 + l16]    = (_Float16)ck[r];
        smem[VT_OFF + (nt * 16 + l16) * N_SEQ + row] = (_Float16)(cv[r] + bias);
      }
    }
  }
  __syncthreads();

  // =========================================================================
  // Phase B: flash attention. Wave w handles i-tiles w, w+8, w+16, w+24.
  // =========================================================================
#pragma unroll 1
  for (int it = 0; it < 4; ++it) {
    const int i0 = (wave + it * 8) * 16;

    // ---- Q tile: X rows i0..i0+15 @ Wq, converted to 4 A-fragments ----
    v16h xa[4];
#pragma unroll
    for (int kc = 0; kc < 4; ++kc)
      xa[kc] = ld_a_x(xn + (size_t)(i0 + l16) * D_DIM + kc * 32, hh);

    v16h qa[4];
#pragma unroll 1
    for (int c = 0; c < 4; ++c) {
#pragma unroll
      for (int s = 0; s < 2; ++s) {
        const int nt = 2 * c + s;
        v8f q = {};
#pragma unroll
        for (int kc = 0; kc < 4; ++kc) {
          // Wq^T rows are contiguous in k -> vector ds loads
          const v16h bq = ld_frag_lds(WQT_OFF + (nt * 16 + l16) * D_DIM +
                                      kc * 32 + hh * 16);
          q = wmma_f16(xa[kc], bq, q);
        }
#pragma unroll
        for (int r = 0; r < 8; ++r)
          smem[stg + (hh * 8 + r) * 32 + s * 16 + l16] = (_Float16)q[r];
      }
      // C-layout -> A-layout via per-wave stage (same-wave DS is in-order)
      qa[c] = ld_a_stage(stg, l16, hh);
    }

    // ---- online softmax over j ----
    v8f o[8];
#pragma unroll
    for (int ntd = 0; ntd < 8; ++ntd) { v8f z = {}; o[ntd] = z; }
    float mrow[8], lrow[8];
#pragma unroll
    for (int r = 0; r < 8; ++r) { mrow[r] = -3.0e38f; lrow[r] = 0.f; }

#pragma unroll 1
    for (int j0 = 0; j0 < N_SEQ; j0 += 32) {
      // S tiles: rows i0..i0+15, cols j0..j0+31 (contiguous B-frags in K_s)
      v8f s0 = {}, s1 = {};
#pragma unroll
      for (int kc = 0; kc < 4; ++kc) {
        const v16h b0 = ld_frag_lds(K_OFF + (j0 + l16) * D_DIM +
                                    kc * 32 + hh * 16);
        const v16h b1 = ld_frag_lds(K_OFF + (j0 + 16 + l16) * D_DIM +
                                    kc * 32 + hh * 16);
        s0 = wmma_f16(qa[kc], b0, s0);
        s1 = wmma_f16(qa[kc], b1, s1);
      }
      // per-row stats: row m = hh*8+r spans the 16 lanes of this half
#pragma unroll
      for (int r = 0; r < 8; ++r) {
        float rm = fmaxf(s0[r], s1[r]);
        rm = fmaxf(rm, __shfl_xor(rm, 8, 32));
        rm = fmaxf(rm, __shfl_xor(rm, 4, 32));
        rm = fmaxf(rm, __shfl_xor(rm, 2, 32));
        rm = fmaxf(rm, __shfl_xor(rm, 1, 32));
        const float mnew = fmaxf(mrow[r], rm);
        const float corr = __expf(mrow[r] - mnew);
        const float p0 = __expf(s0[r] - mnew);
        const float p1 = __expf(s1[r] - mnew);
        float rs = p0 + p1;
        rs += __shfl_xor(rs, 8, 32);
        rs += __shfl_xor(rs, 4, 32);
        rs += __shfl_xor(rs, 2, 32);
        rs += __shfl_xor(rs, 1, 32);
        lrow[r] = lrow[r] * corr + rs;
        mrow[r] = mnew;
#pragma unroll
        for (int ntd = 0; ntd < 8; ++ntd) o[ntd][r] *= corr;
        smem[stg + (hh * 8 + r) * 32 + l16]      = (_Float16)p0;
        smem[stg + (hh * 8 + r) * 32 + 16 + l16] = (_Float16)p1;
      }
      // P (16x32) as A-fragment
      const v16h pa = ld_a_stage(stg, l16, hh);
      // O += P @ V  (V^T rows contiguous in j -> vector ds loads)
#pragma unroll
      for (int ntd = 0; ntd < 8; ++ntd) {
        const v16h bvf = ld_frag_lds(VT_OFF + (ntd * 16 + l16) * N_SEQ +
                                     j0 + hh * 16);
        o[ntd] = wmma_f16(pa, bvf, o[ntd]);
      }
    }

    // ---- normalize rows and store out[n][i][d] (f32) ----
#pragma unroll
    for (int r = 0; r < 8; ++r) {
      const float inv = 1.0f / lrow[r];
      const int row = i0 + hh * 8 + r;
      float* op = out + ((size_t)n * N_SEQ + row) * D_DIM;
#pragma unroll
      for (int ntd = 0; ntd < 8; ++ntd)
        op[ntd * 16 + l16] = o[ntd][r] * inv;
    }
  }
}

extern "C" void kernel_launch(void* const* d_in, const int* in_sizes, int n_in,
                              void* d_out, int out_size, void* d_ws, size_t ws_size,
                              hipStream_t stream) {
  const float* x    = (const float*)d_in[0];   // [1,512,512,128]
  const float* w_qk = (const float*)d_in[1];   // [128,256]
  const float* w_v  = (const float*)d_in[2];   // [128,128]
  const float* b_v  = (const float*)d_in[3];   // [128]
  float* out = (float*)d_out;                  // [1,512,512,128]

  const size_t shmem = (size_t)SMEM_HALFS * sizeof(_Float16);  // 296 KB

  hipLaunchKernelGGL(pairwise_row_attention_kernel, dim3(N_SEQ), dim3(256),
                     shmem, stream, x, w_qk, w_v, b_v, out);
}
